// Mamba_dflow_MAE_60309930771131
// MI455X (gfx1250) — matrix operand order repaired and checked by reading, not compiled, not run on hardware
//
#include <hip/hip_runtime.h>

// ---------------- problem constants ----------------
constexpr int B   = 4;
constexpr int L   = 1024;
constexpr int H   = 256;
constexpr int DI  = 512;
constexpr int DS  = 16;
constexpr int DC  = 4;
constexpr int NL  = 2;
constexpr int M   = B * L;          // 4096 tokens
constexpr int BLK = 64;
constexpr int NLAB = 4;

typedef __attribute__((ext_vector_type(16))) _Float16 v16h;
typedef __attribute__((ext_vector_type(8)))  float    v8f;

#if __has_builtin(__builtin_amdgcn_global_load_async_to_lds_b32) && \
    __has_builtin(__builtin_amdgcn_s_wait_asynccnt)
#define USE_ASYNC_LDS 1
typedef __attribute__((address_space(1))) int as1_int;
typedef __attribute__((address_space(3))) int as3_int;
#endif

// ---------------- f32 -> f16 conversion ----------------
__global__ void k_f32_to_f16(const float* __restrict__ src, _Float16* __restrict__ dst, int n) {
  int i = blockIdx.x * blockDim.x + threadIdx.x;
  if (i < n) dst[i] = (_Float16)src[i];
}

// pad dt weight (rows x 16) -> (rows x 32), zeros in cols 16..31
__global__ void k_dtw_pad(const float* __restrict__ src, _Float16* __restrict__ dst, int rows) {
  int i = blockIdx.x * blockDim.x + threadIdx.x;
  if (i >= rows * 32) return;
  int r = i >> 5, c = i & 31;
  dst[i] = (c < 16) ? (_Float16)src[r * 16 + c] : (_Float16)0.0f;
}

// ---------------- embedding + LN + gate ----------------
__global__ void k_embed(const int* __restrict__ ids, const float* __restrict__ gate,
                        const float* __restrict__ wemb, const float* __restrict__ pemb,
                        const float* __restrict__ pemb2, const float* __restrict__ ttemb,
                        const float* __restrict__ lnw, const float* __restrict__ lnb,
                        float* __restrict__ h) {
  int m = blockIdx.x;        // token
  int c = threadIdx.x;       // 0..255
  int l = m & (L - 1);
  int id = ids[m];
  float we = wemb[(size_t)id * H + c] + ttemb[c];
  float e  = pemb[l * H + c] * we + pemb2[l * H + c];
  __shared__ float red[H];
  red[c] = e; __syncthreads();
  for (int s = H / 2; s > 0; s >>= 1) { if (c < s) red[c] += red[c + s]; __syncthreads(); }
  float mean = red[0] * (1.0f / H); __syncthreads();
  float d = e - mean;
  red[c] = d * d; __syncthreads();
  for (int s = H / 2; s > 0; s >>= 1) { if (c < s) red[c] += red[c + s]; __syncthreads(); }
  float var = red[0] * (1.0f / H);
  float nv  = d * rsqrtf(var + 1e-12f) * lnw[c] + lnb[c];
  h[(size_t)m * H + c] = gate[m] * nv;
}

// ---------------- LN over H (residual sum -> h) ----------------
__global__ void k_ln(const float* __restrict__ src, const float* __restrict__ lnw,
                     const float* __restrict__ lnb, float* __restrict__ dst) {
  int m = blockIdx.x;
  int c = threadIdx.x;
  float e = src[(size_t)m * H + c];
  __shared__ float red[H];
  red[c] = e; __syncthreads();
  for (int s = H / 2; s > 0; s >>= 1) { if (c < s) red[c] += red[c + s]; __syncthreads(); }
  float mean = red[0] * (1.0f / H); __syncthreads();
  float d = e - mean;
  red[c] = d * d; __syncthreads();
  for (int s = H / 2; s > 0; s >>= 1) { if (c < s) red[c] += red[c + s]; __syncthreads(); }
  float var = red[0] * (1.0f / H);
  dst[(size_t)m * H + c] = d * rsqrtf(var + 1e-12f) * lnw[c] + lnb[c];
}

// ---------------- pack h (f32, optionally L-flipped per sequence) to f16 ----------------
__global__ void k_pack_f16(const float* __restrict__ src, _Float16* __restrict__ dst, int flip) {
  int idx = blockIdx.x * blockDim.x + threadIdx.x;
  if (idx >= M * H) return;
  int c = idx & (H - 1);
  int m = idx >> 8;
  int l = m & (L - 1);
  int sm = flip ? (m - l + (L - 1 - l)) : m;
  dst[idx] = (_Float16)src[(size_t)sm * H + c];
}

// ---------------- WMMA GEMM: C[M,N] = A[M,K] * W[N,K]^T  (+optional addsrc) ----------
// One wave computes a 16 x (16*NT) output block: A fragment loaded once, reused for NT
// back-to-back v_wmma ops (A traffic /NT, deeper XDL pipelining).
// grid = (N/(16*NT), M/16), block = 32; EXEC all ones for WMMA.
template <int NT>
__global__ void k_gemm_nt(const _Float16* __restrict__ A, const _Float16* __restrict__ W,
                          float* __restrict__ C, int N, int K,
                          const float* __restrict__ addsrc) {
  int lane = threadIdx.x;           // 0..31
  int mrow = lane & 15;             // A row within tile / C column index
  int half = lane >> 4;             // half-wave select
  int m0 = blockIdx.y * 16, n0 = blockIdx.x * 16 * NT;
  v8f acc[NT];
#pragma unroll
  for (int t = 0; t < NT; ++t) acc[t] = {};
  const _Float16* arow = A + (size_t)(m0 + mrow) * K;
  for (int k0 = 0; k0 < K; k0 += 32) {
    v16h a;
#pragma unroll
    for (int v = 0; v < 8; ++v) {
      // 16-bit A 16x32 layout: lanes0-15 v0..3 K=0..7, v4..7 K=16..23; lanes16-31 +8
      int kb = ((v < 4) ? 0 : 16) + half * 8 + (v & 3) * 2;
      a[2 * v]     = arow[k0 + kb];
      a[2 * v + 1] = arow[k0 + kb + 1];
    }
    if (k0 + 32 < K) __builtin_prefetch(arow + k0 + 32, 0, 1);  // global_prefetch_b8
#pragma unroll
    for (int t = 0; t < NT; ++t) {
      // B 32x16: n = lane&15, lanes0-15 K=0..7,16..23; lanes16-31 +8
      const _Float16* wp = W + (size_t)(n0 + t * 16 + mrow) * K + k0 + half * 8;
      v16h b;
#pragma unroll
      for (int i = 0; i < 16; ++i) b[i] = wp[(i < 8) ? i : i + 8];
      if (k0 + 32 < K) __builtin_prefetch(wp + 32, 0, 1);
      acc[t] = __builtin_amdgcn_wmma_f32_16x16x32_f16(false, a, false, b, (short)0, acc[t],
                                                      false, false);
    }
  }
#pragma unroll
  for (int t = 0; t < NT; ++t) {
    int col = n0 + t * 16 + mrow;
#pragma unroll
    for (int r = 0; r < 8; ++r) {                 // C layout: M = half*8 + r, N = lane&15
      int row = m0 + half * 8 + r;
      float v = acc[t][r];
      if (addsrc) v += addsrc[(size_t)row * N + col];
      C[(size_t)row * N + col] = v;
    }
  }
}

// ---------------- causal depthwise conv (DC=4) + SiLU ----------------
__global__ void k_conv_silu(const float* __restrict__ xz, const float* __restrict__ cw,
                            const float* __restrict__ cb, float* __restrict__ xcf,
                            _Float16* __restrict__ xc16) {
  int idx = blockIdx.x * blockDim.x + threadIdx.x;  // over M*DI
  if (idx >= M * DI) return;
  int d = idx & (DI - 1);
  int m = idx >> 9;
  int l = m & (L - 1);
  float acc = cb[d];
#pragma unroll
  for (int k = 0; k < DC; ++k) {
    int li = l + k - (DC - 1);
    if (li >= 0) acc += cw[d * DC + k] * xz[(size_t)(m + k - (DC - 1)) * (2 * DI) + d];
  }
  float s = acc / (1.0f + __expf(-acc));
  xcf[idx]  = s;
  xc16[idx] = (_Float16)s;
}

// ---------------- dt slice (cols 0..15 of xdbl) -> padded f16 (M x 32) ----------------
__global__ void k_split_dt(const float* __restrict__ xdbl, _Float16* __restrict__ dt16) {
  int idx = blockIdx.x * blockDim.x + threadIdx.x;  // over M*32
  if (idx >= M * 32) return;
  int m = idx >> 5, c = idx & 31;
  dt16[idx] = (c < 16) ? (_Float16)xdbl[(size_t)m * 48 + c] : (_Float16)0.0f;
}

// ---------------- dt = softplus(dt_raw + bias) in place ----------------
__global__ void k_bias_softplus(float* __restrict__ dtf, const float* __restrict__ bias) {
  int idx = blockIdx.x * blockDim.x + threadIdx.x;  // over M*DI
  if (idx >= M * DI) return;
  int d = idx & (DI - 1);
  float x = dtf[idx] + bias[d];
  dtf[idx] = (x > 20.0f) ? x : log1pf(__expf(x));
}

// ---------------- selective scan, fused gating: y = (scan + x*D) * silu(z) ------------
// grid = B blocks, block = DI threads; B/C broadcast staged through LDS per timestep,
// via GLOBAL_LOAD_ASYNC_TO_LDS_B32 when the toolchain exposes it (ASYNCcnt path).
__global__ void k_scan(const float* __restrict__ xz, const float* __restrict__ xcf,
                       const float* __restrict__ xdbl, const float* __restrict__ dtf,
                       const float* __restrict__ Alog, const float* __restrict__ Dp,
                       _Float16* __restrict__ y16) {
  int b = blockIdx.x;
  int d = threadIdx.x;            // 0..DI-1
  float A[DS], hs[DS];
#pragma unroll
  for (int s = 0; s < DS; ++s) { A[s] = -__expf(Alog[d * DS + s]); hs[s] = 0.0f; }
  float Dd = Dp[d];
  __shared__ float sBC[2 * DS];
  for (int l = 0; l < L; ++l) {
    int m = b * L + l;
    __syncthreads();
    if (d < 2 * DS) {
      const float* src = xdbl + (size_t)m * 48 + 16 + d;
#if defined(USE_ASYNC_LDS)
      __builtin_amdgcn_global_load_async_to_lds_b32(
          (as1_int*)src, (as3_int*)&sBC[d], 0, 0);
#else
      sBC[d] = *src;
#endif
    }
#if defined(USE_ASYNC_LDS)
    __builtin_amdgcn_s_wait_asynccnt(0);
#endif
    __syncthreads();
    float dt = dtf[(size_t)m * DI + d];
    float x  = xcf[(size_t)m * DI + d];
    float y  = 0.0f;
#pragma unroll
    for (int s = 0; s < DS; ++s) {
      float dA = __expf(dt * A[s]);
      hs[s] = dA * hs[s] + dt * sBC[s] * x;
      y += hs[s] * sBC[DS + s];
    }
    float z  = xz[(size_t)m * (2 * DI) + DI + d];
    float yv = (y + x * Dd) * (z / (1.0f + __expf(-z)));
    y16[(size_t)m * DI + d] = (_Float16)yv;
  }
}

// ---------------- sumbuf += flip_L(dirout) ----------------
__global__ void k_add_flip(float* __restrict__ sumbuf, const float* __restrict__ dirout) {
  int idx = blockIdx.x * blockDim.x + threadIdx.x;  // over M*H
  if (idx >= M * H) return;
  int c = idx & (H - 1);
  int m = idx >> 8;
  int l = m & (L - 1);
  int sm = m - l + (L - 1 - l);
  sumbuf[idx] += dirout[(size_t)sm * H + c];
}

// ---------------- max over L of h*gate ----------------
__global__ void k_maxpool(const float* __restrict__ h, const float* __restrict__ gate,
                          float* __restrict__ pooled) {
  int idx = blockIdx.x * blockDim.x + threadIdx.x;  // over B*H
  if (idx >= B * H) return;
  int b = idx >> 8, c = idx & (H - 1);
  float mx = -3.4e38f;
  for (int l = 0; l < L; ++l) {
    float v = h[(size_t)(b * L + l) * H + c] * gate[b * L + l];
    mx = fmaxf(mx, v);
  }
  pooled[idx] = mx;
}

// ---------------- tiny head GEMV: out[r,c] = bias[c] + sum_k in[r,k]*W[c,k] ----------
__global__ void k_head(const float* __restrict__ in, const float* __restrict__ Wt,
                       const float* __restrict__ bias, float* __restrict__ out,
                       int rows, int N, int K) {
  int idx = blockIdx.x * blockDim.x + threadIdx.x;
  if (idx >= rows * N) return;
  int r = idx / N, c = idx - r * N;
  float acc = bias[c];
  for (int k = 0; k < K; ++k) acc += in[(size_t)r * K + k] * Wt[(size_t)c * K + k];
  out[idx] = acc;
}

__global__ void k_copy(const float* __restrict__ src, float* __restrict__ dst, int n) {
  int i = blockIdx.x * blockDim.x + threadIdx.x;
  if (i < n) dst[i] = src[i];
}

// =====================================================================================
extern "C" void kernel_launch(void* const* d_in, const int* in_sizes, int n_in,
                              void* d_out, int out_size, void* d_ws, size_t ws_size,
                              hipStream_t stream) {
  (void)in_sizes; (void)n_in; (void)out_size; (void)ws_size;
  const int*   input_ids = (const int*)  d_in[0];
  const float* gate      = (const float*)d_in[1];
  const float* word_emb  = (const float*)d_in[2];
  const float* pos_emb   = (const float*)d_in[3];
  const float* pos_emb2  = (const float*)d_in[4];
  const float* tt_emb    = (const float*)d_in[5];
  const float* emb_ln_w  = (const float*)d_in[6];
  const float* emb_ln_b  = (const float*)d_in[7];
  const float* m_in_w    = (const float*)d_in[8];
  const float* m_conv_w  = (const float*)d_in[9];
  const float* m_conv_b  = (const float*)d_in[10];
  const float* m_x_w     = (const float*)d_in[11];
  const float* m_dt_w    = (const float*)d_in[12];
  const float* m_dt_b    = (const float*)d_in[13];
  const float* m_Alog    = (const float*)d_in[14];
  const float* m_D       = (const float*)d_in[15];
  const float* m_out_w   = (const float*)d_in[16];
  const float* blk_ln_w  = (const float*)d_in[17];
  const float* blk_ln_b  = (const float*)d_in[18];
  const float* d2_w      = (const float*)d_in[19];
  const float* d2_b      = (const float*)d_in[20];
  const float* op_w      = (const float*)d_in[21];
  const float* op_b      = (const float*)d_in[22];
  float* out = (float*)d_out;

  // ------------- workspace carve-up -------------
  char* ws = (char*)d_ws;
  size_t off = 0;
  auto alloc = [&](size_t bytes) -> void* {
    void* p = ws + off;
    off = (off + bytes + 255) & ~(size_t)255;
    return p;
  };
  float*    h       = (float*)   alloc((size_t)M * H * 4);
  _Float16* hf16    = (_Float16*)alloc((size_t)M * H * 2);
  float*    xz      = (float*)   alloc((size_t)M * 2 * DI * 4);
  float*    xcf     = (float*)   alloc((size_t)M * DI * 4);
  _Float16* xc16    = (_Float16*)alloc((size_t)M * DI * 2);
  float*    xdbl    = (float*)   alloc((size_t)M * 48 * 4);
  _Float16* dt16    = (_Float16*)alloc((size_t)M * 32 * 2);
  float*    dtf     = (float*)   alloc((size_t)M * DI * 4);
  _Float16* y16     = (_Float16*)alloc((size_t)M * DI * 2);
  float*    dirout  = (float*)   alloc((size_t)M * H * 4);
  float*    sumbuf  = (float*)   alloc((size_t)M * H * 4);
  float*    pooled  = (float*)   alloc((size_t)B * H * 4);
  float*    feature = (float*)   alloc((size_t)B * H * 4);
  _Float16* w_in16  = (_Float16*)alloc((size_t)NL * 2 * 2 * DI * H * 2);
  _Float16* w_x16   = (_Float16*)alloc((size_t)NL * 2 * 48 * DI * 2);
  _Float16* w_dt16  = (_Float16*)alloc((size_t)NL * 2 * DI * 32 * 2);
  _Float16* w_out16 = (_Float16*)alloc((size_t)NL * 2 * H * DI * 2);

  auto cdiv = [](int a, int b) { return (a + b - 1) / b; };

  // ------------- weights -> f16 (once per launch) -------------
  {
    int n;
    n = NL * 2 * 2 * DI * H;
    k_f32_to_f16<<<cdiv(n, 256), 256, 0, stream>>>(m_in_w, w_in16, n);
    n = NL * 2 * 48 * DI;
    k_f32_to_f16<<<cdiv(n, 256), 256, 0, stream>>>(m_x_w, w_x16, n);
    n = NL * 2 * H * DI;
    k_f32_to_f16<<<cdiv(n, 256), 256, 0, stream>>>(m_out_w, w_out16, n);
    int rows = NL * 2 * DI;
    k_dtw_pad<<<cdiv(rows * 32, 256), 256, 0, stream>>>(m_dt_w, w_dt16, rows);
  }

  // ------------- embedding -------------
  k_embed<<<M, H, 0, stream>>>(input_ids, gate, word_emb, pos_emb, pos_emb2, tt_emb,
                               emb_ln_w, emb_ln_b, h);

  // ------------- mamba layers -------------
  for (int l = 0; l < NL; ++l) {
    for (int dir = 0; dir < 2; ++dir) {
      int p = l * 2 + dir;
      const _Float16* Wi  = w_in16  + (size_t)p * 2 * DI * H;
      const float*    cw  = m_conv_w + (size_t)p * DI * DC;
      const float*    cb  = m_conv_b + (size_t)p * DI;
      const _Float16* Wx  = w_x16   + (size_t)p * 48 * DI;
      const _Float16* Wdt = w_dt16  + (size_t)p * DI * 32;
      const float*    bdt = m_dt_b  + (size_t)p * DI;
      const float*    Al  = m_Alog  + (size_t)p * DI * DS;
      const float*    Dp  = m_D     + (size_t)p * DI;
      const _Float16* Wo  = w_out16 + (size_t)p * H * DI;

      // input (flipped for bwd) -> f16
      k_pack_f16<<<cdiv(M * H, 256), 256, 0, stream>>>(h, hf16, dir);
      // xz = x @ Wi^T  (M x 1024, K=256): 16x64 per wave
      k_gemm_nt<4><<<dim3((2 * DI) / 64, M / 16), 32, 0, stream>>>(hf16, Wi, xz, 2 * DI, H,
                                                                   nullptr);
      // causal depthwise conv + silu
      k_conv_silu<<<cdiv(M * DI, 256), 256, 0, stream>>>(xz, cw, cb, xcf, xc16);
      // xdbl = xc @ Wx^T (M x 48, K=512): 16x48 per wave
      k_gemm_nt<3><<<dim3(1, M / 16), 32, 0, stream>>>(xc16, Wx, xdbl, 48, DI, nullptr);
      // dt slice -> padded f16, dt_raw = dt @ Wdt^T (M x 512, K=32)
      k_split_dt<<<cdiv(M * 32, 256), 256, 0, stream>>>(xdbl, dt16);
      k_gemm_nt<4><<<dim3(DI / 64, M / 16), 32, 0, stream>>>(dt16, Wdt, dtf, DI, 32, nullptr);
      k_bias_softplus<<<cdiv(M * DI, 256), 256, 0, stream>>>(dtf, bdt);
      // selective scan + gating
      k_scan<<<B, DI, 0, stream>>>(xz, xcf, xdbl, dtf, Al, Dp, y16);
      // out-proj (M x 256, K=512); fwd fuses +h residual, bwd adds flipped
      if (dir == 0) {
        k_gemm_nt<4><<<dim3(H / 64, M / 16), 32, 0, stream>>>(y16, Wo, sumbuf, H, DI, h);
      } else {
        k_gemm_nt<4><<<dim3(H / 64, M / 16), 32, 0, stream>>>(y16, Wo, dirout, H, DI, nullptr);
        k_add_flip<<<cdiv(M * H, 256), 256, 0, stream>>>(sumbuf, dirout);
      }
    }
    // h = LN(f + b + h)
    k_ln<<<M, H, 0, stream>>>(sumbuf, blk_ln_w + (size_t)l * H, blk_ln_b + (size_t)l * H, h);
  }

  // ------------- classification head -------------
  k_maxpool<<<cdiv(B * H, 256), 256, 0, stream>>>(h, gate, pooled);
  k_head<<<cdiv(B * H, 256), 256, 0, stream>>>(pooled, d2_w, d2_b, feature, B, H, H);
  // logits straight into d_out[0 .. 1023]  ((B,64,4) row-major == (B,256))
  k_head<<<cdiv(B * BLK * NLAB, 256), 256, 0, stream>>>(feature, op_w, op_b, out, B, BLK * NLAB, H);
  // gate -> d_out[1024 .. 5119], feature -> d_out[5120 .. 6143]
  k_copy<<<cdiv(B * L, 256), 256, 0, stream>>>(gate, out + B * BLK * NLAB, B * L);
  k_copy<<<cdiv(B * H, 256), 256, 0, stream>>>(feature, out + B * BLK * NLAB + B * L, B * H);
}
